// FourierCrossAttention_67637144978395
// MI455X (gfx1250) — matrix operand order repaired and checked by reading, not compiled, not run on hardware
//
#include <hip/hip_runtime.h>

// ---------------------------------------------------------------------------
// FourierCrossAttention for MI455X (gfx1250, wave32, WMMA)
// B=8, L=4096, H=8, E=64, MODES=64
// ---------------------------------------------------------------------------

typedef __attribute__((ext_vector_type(16))) _Float16 v16h;
typedef __attribute__((ext_vector_type(8)))  _Float16 v8h;
typedef __attribute__((ext_vector_type(8)))  float    v8f;

constexpr int Bc = 8;
constexpr int Lc = 4096;
constexpr int Hc = 8;
constexpr int Ec = 64;
constexpr int Mc = 64;       // modes
constexpr int Jc = 2 * Mc;   // 128 = interleaved re/im columns
constexpr int BH = Bc * Hc;  // 64
constexpr int HE = Hc * Ec;  // 512

// ---------------------------------------------------------------------------
// Kernel 0: build f16 twiddle tables.
//   Wf[l][2m]   =  cos(2*pi*l*idx[m]/L)        (rfft real part basis)
//   Wf[l][2m+1] = -sin(2*pi*l*idx[m]/L)        (rfft imag part basis)
//   Wi[2m][l]   =  cm*cos(...), Wi[2m+1][l] = -cm*sin(...)
//   cm = 1 if idx==0 else 2   (irfft Hermitian fold; 1/L applied later)
// ---------------------------------------------------------------------------
__global__ __launch_bounds__(256)
void k_twiddle(const int* __restrict__ idxq, const int* __restrict__ idxkv,
               _Float16* __restrict__ Wfq, _Float16* __restrict__ Wfkv,
               _Float16* __restrict__ Wi) {
  const int t = blockIdx.x * 256 + threadIdx.x;   // 0 .. 262143
  const float w0 = 6.28318530717958647693f / (float)Lc;
  if (blockIdx.y < 2) {
    const int* idx = (blockIdx.y == 0) ? idxq : idxkv;
    _Float16* Wf   = (blockIdx.y == 0) ? Wfq  : Wfkv;
    const int m = t & 63;
    const int l = t >> 6;                          // 0..4095
    const int r = (l * idx[m]) & (Lc - 1);         // exact reduction mod L
    float s, c;
    __sincosf((float)r * w0, &s, &c);
    Wf[l * Jc + 2 * m]     = (_Float16)c;
    Wf[l * Jc + 2 * m + 1] = (_Float16)(-s);
  } else {
    const int l  = t & (Lc - 1);
    const int m  = t >> 12;                        // 0..63
    const int id = idxq[m];
    const int r  = (l * id) & (Lc - 1);
    float s, c;
    __sincosf((float)r * w0, &s, &c);
    const float cm = (id == 0) ? 1.f : 2.f;
    Wi[(2 * m) * Lc + l]     = (_Float16)(cm * c);
    Wi[(2 * m + 1) * Lc + l] = (_Float16)(-cm * s);
  }
}

// ---------------------------------------------------------------------------
// Kernel 1: sparse forward DFT as GEMM.
//   X[e, j] = sum_l x[b,l,h,e] * Wf[l, j]     (E=64 x J=128, K=L=4096)
// Block = one (b,h,tensor). 8 waves; wave w owns n-tile j=[16w,16w+16),
// all 4 m-tiles. x is double-buffered through LDS as f16: the next chunk's
// global_load_b128s are issued before the current chunk's WMMA chain so the
// HBM latency hides behind 16 wmmas/chunk.
// ---------------------------------------------------------------------------
constexpr int ROWH = 136;   // padded LDS row (halves): 272B = 16B-aligned

__global__ __launch_bounds__(256)
void k_dft(const float* __restrict__ q, const float* __restrict__ k,
           const float* __restrict__ v,
           const _Float16* __restrict__ Wfq, const _Float16* __restrict__ Wfkv,
           float* __restrict__ Xq, float* __restrict__ Xk,
           float* __restrict__ Xv) {
  __shared__ __align__(16) _Float16 lds[2][64 * ROWH];   // 2 x 17 KB
  const int bh  = blockIdx.x;
  const int b   = bh >> 3, h = bh & 7;
  const int ten = blockIdx.y;
  const float*    src = (ten == 0) ? q   : (ten == 1) ? k   : v;
  const _Float16* Wf  = (ten == 0) ? Wfq : Wfkv;
  float*          dst = (ten == 0) ? Xq  : (ten == 1) ? Xk  : Xv;

  const int t = threadIdx.x, lane = t & 31, w = t >> 5;
  const size_t base = (size_t)b * Lc * HE + (size_t)h * Ec;

  const int e4     = (t & 15) * 4;   // staging: 4 consecutive e per thread
  const int lpBase = t >> 4;         // 0..15

  auto stageLoad = [&](int c, float4* r) {
    const int l0 = c * 128;
    for (int i = 0; i < 8; ++i) {
      const int lp = lpBase + i * 16;
      r[i] = *(const float4*)(src + base + (size_t)(l0 + lp) * HE + e4);
    }
  };
  auto stageStore = [&](_Float16* buf, const float4* r) {
    for (int i = 0; i < 8; ++i) {
      const int lp = lpBase + i * 16;
      buf[(e4 + 0) * ROWH + lp] = (_Float16)r[i].x;
      buf[(e4 + 1) * ROWH + lp] = (_Float16)r[i].y;
      buf[(e4 + 2) * ROWH + lp] = (_Float16)r[i].z;
      buf[(e4 + 3) * ROWH + lp] = (_Float16)r[i].w;
    }
  };

  v8f acc[4] = {};
  float4 r[8];
  stageLoad(0, r);
  stageStore(lds[0], r);

  for (int c = 0; c < 32; ++c) {
    __syncthreads();   // buf[c&1] staged; buf[(c+1)&1] free
    if (c + 1 < 32) stageLoad(c + 1, r);          // overlap with WMMAs below
    if (c + 2 < 32)  // global_prefetch_b8: pull chunk c+2 lines toward L2
      __builtin_prefetch(src + base + (size_t)((c + 2) * 128 + lpBase) * HE + e4, 0, 1);

    const _Float16* buf = lds[c & 1];
    for (int s = 0; s < 4; ++s) {
      // B fragment: lane holds column j = 16w + lane%16; halves 0-7 = K kb0.., 8-15 = +16
      const int kb0 = s * 32 + ((lane >> 4) << 3);
      const int ncol = w * 16 + (lane & 15);
      v16h bf;
      for (int i = 0; i < 16; ++i)
        bf[i] = Wf[(size_t)(c * 128 + kb0 + (i & 7) + ((i >> 3) << 4)) * Jc + ncol];
      v16h af[4];
      for (int m = 0; m < 4; ++m) {
        // A fragment: row e = 16m + lane%16; halves 0-7 = K kb0.., 8-15 = +16
        const _Float16* ap = buf + (m * 16 + (lane & 15)) * ROWH + kb0;
        const v8h a0 = *(const v8h*)ap;
        const v8h a1 = *(const v8h*)(ap + 16);
        af[m] = __builtin_shufflevector(
            a0, a1, 0, 1, 2, 3, 4, 5, 6, 7, 8, 9, 10, 11, 12, 13, 14, 15);
      }
      for (int m = 0; m < 4; ++m)
        acc[m] = __builtin_amdgcn_wmma_f32_16x16x32_f16(
            false, af[m], false, bf, (short)0, acc[m], false, false);
    }
    if (c + 1 < 32) stageStore(lds[(c + 1) & 1], r);
  }

  // ---- epilogue: C layout -> X[bh][e][j] f32 ------------------------------
  float* o = dst + (size_t)bh * (Ec * Jc);
  const int n0  = w * 16 + (lane & 15);
  const int mhi = (lane >> 4) * 8;
  for (int m = 0; m < 4; ++m)
    for (int rr = 0; rr < 8; ++rr)
      o[(m * 16 + mhi + rr) * Jc + n0] = acc[m][rr];
}

// ---------------------------------------------------------------------------
// Kernel 2: frequency-domain complex attention (small: f32 VALU, f16 LDS).
//   S[x,y]    = sum_e Q[e,x]*K[e,y]            (complex, no conj)
//   T         = ctanh(S)
//   U[e,x]    = sum_y T[x,y]*V[e,y]
//   Xw[o,x]   = sum_e U[e,x]*(wr+i*wi)[h,e,o,x]   -> f16, A-matrix for iDFT
// ---------------------------------------------------------------------------
__global__ __launch_bounds__(256)
void k_mid(const float* __restrict__ Xq, const float* __restrict__ Xk,
           const float* __restrict__ Xv, const float* __restrict__ gwr,
           const float* __restrict__ gwi, _Float16* __restrict__ Xw) {
  __shared__ _Float16 lQ[Ec * Jc];   // Q spectra, later reused for U
  __shared__ _Float16 lK[Ec * Jc];
  __shared__ _Float16 lV[Ec * Jc];
  __shared__ _Float16 lS[Mc * Jc];   // tanh(S)
  const int bh = blockIdx.x;
  const int h  = bh & 7;
  const int t  = threadIdx.x;

  const float* gq = Xq + (size_t)bh * (Ec * Jc);
  const float* gk = Xk + (size_t)bh * (Ec * Jc);
  const float* gv = Xv + (size_t)bh * (Ec * Jc);
  for (int i = 0; i < 32; ++i) {
    const int idx = t + i * 256;
    lQ[idx] = (_Float16)gq[idx];
    lK[idx] = (_Float16)gk[idx];
    lV[idx] = (_Float16)gv[idx];
  }
  __syncthreads();

  // ---- S = Q^T K, then complex tanh --------------------------------------
  for (int i = 0; i < 16; ++i) {
    const int idx = t + i * 256;
    const int x = idx >> 6, y = idx & 63;
    float ar = 0.f, ai = 0.f;
    for (int e = 0; e < 64; ++e) {
      const float qr = (float)lQ[e * Jc + 2 * x];
      const float qi = (float)lQ[e * Jc + 2 * x + 1];
      const float kr = (float)lK[e * Jc + 2 * y];
      const float ki = (float)lK[e * Jc + 2 * y + 1];
      ar += qr * kr - qi * ki;
      ai += qr * ki + qi * kr;
    }
    // tanh(a+bi) = (sinh2a + i sin2b) / (cosh2a + cos2b), saturate for big |a|
    const float a2 = 2.f * ar, b2 = 2.f * ai;
    float tr, ti;
    if (fabsf(a2) > 20.f) {
      tr = (a2 > 0.f) ? 1.f : -1.f;
      ti = 0.f;
    } else {
      const float ex = __expf(a2), exm = 1.f / ex;
      const float sh = 0.5f * (ex - exm), ch = 0.5f * (ex + exm);
      float sb, cb;
      __sincosf(b2, &sb, &cb);
      const float d = ch + cb;
      tr = sh / d;
      ti = sb / d;
    }
    lS[x * Jc + 2 * y]     = (_Float16)tr;
    lS[x * Jc + 2 * y + 1] = (_Float16)ti;
  }
  __syncthreads();

  // ---- U[e,x] = sum_y T[x,y] * V[e,y]  (write into lQ, Q is dead) --------
  for (int i = 0; i < 16; ++i) {
    const int idx = t + i * 256;
    const int e = idx >> 6, x = idx & 63;
    float ar = 0.f, ai = 0.f;
    for (int y = 0; y < 64; ++y) {
      const float sr = (float)lS[x * Jc + 2 * y];
      const float si = (float)lS[x * Jc + 2 * y + 1];
      const float vr = (float)lV[e * Jc + 2 * y];
      const float vi = (float)lV[e * Jc + 2 * y + 1];
      ar += sr * vr - si * vi;
      ai += sr * vi + si * vr;
    }
    lQ[e * Jc + 2 * x]     = (_Float16)ar;
    lQ[e * Jc + 2 * x + 1] = (_Float16)ai;
  }
  __syncthreads();

  // ---- per-mode weight contraction over e --------------------------------
  _Float16* out = Xw + (size_t)bh * (Ec * Jc);
  for (int i = 0; i < 16; ++i) {
    const int idx = t + i * 256;
    const int o = idx >> 6, x = idx & 63;
    const size_t wb = (size_t)h * (64 * 64 * 64) + (size_t)o * 64 + x;
    float ar = 0.f, ai = 0.f;
    for (int e = 0; e < 64; ++e) {
      const float wre = gwr[wb + (size_t)e * 4096];
      const float wie = gwi[wb + (size_t)e * 4096];
      const float xr = (float)lQ[e * Jc + 2 * x];
      const float xi = (float)lQ[e * Jc + 2 * x + 1];
      ar += xr * wre - xi * wie;
      ai += xr * wie + xi * wre;
    }
    out[o * Jc + 2 * x]     = (_Float16)ar;
    out[o * Jc + 2 * x + 1] = (_Float16)ai;
  }
}

// ---------------------------------------------------------------------------
// Kernel 3: inverse sparse DFT as GEMM.
//   out[e,l] = SCALE * sum_j Xw[e,j] * Wi[j,l]   (K=128, N tiled by 256)
// grid (16 l-blocks, 64 bh); wave owns 2 n-tiles x 4 m-tiles (frag reuse).
// A-matrix staged with CDNA5 async-to-LDS (ASYNCcnt), no VGPR round-trip.
// SCALE = 1/(IN_C*OUT_C*L) = 2^-30 (2x/1x bin weights folded into Wi).
// ---------------------------------------------------------------------------
__global__ __launch_bounds__(256)
void k_idft(const _Float16* __restrict__ Xw, const _Float16* __restrict__ Wi,
            float* __restrict__ out) {
  __shared__ __align__(16) _Float16 lA[Ec * Jc];   // 16 KB
  const int bh = blockIdx.y;
  const int l0 = blockIdx.x * 256;
  const int t = threadIdx.x, lane = t & 31, w = t >> 5;

  // ---- async stage: 16 KB global f16 -> LDS, b128 per lane per iter ------
  {
    const char* gbase = (const char*)(Xw + (size_t)bh * (Ec * Jc));
    const uint ldsBase = (uint)(uintptr_t)lA;   // low 32 bits = LDS offset
    for (int i = 0; i < 4; ++i) {
      const uint off = (uint)(t + i * 256) * 16u;
      const uint lad = ldsBase + off;
      const void* ga = gbase + off;
      asm volatile("global_load_async_to_lds_b128 %0, %1, off"
                   :: "v"(lad), "v"(ga) : "memory");
    }
    asm volatile("s_wait_asynccnt 0x0" ::: "memory");
  }
  __syncthreads();

  v8f acc[2][4] = {};
  for (int s = 0; s < 4; ++s) {
    const int kb0 = s * 32 + ((lane >> 4) << 3);
    v16h af[4];
    for (int m = 0; m < 4; ++m) {
      const _Float16* ap = lA + (m * 16 + (lane & 15)) * Jc + kb0;
      const v8h a0 = *(const v8h*)ap;
      const v8h a1 = *(const v8h*)(ap + 16);
      af[m] = __builtin_shufflevector(
          a0, a1, 0, 1, 2, 3, 4, 5, 6, 7, 8, 9, 10, 11, 12, 13, 14, 15);
    }
    v16h bf[2];   // B: lane = column l, halves 0-7 = K kb0.., 8-15 = +16
    for (int n = 0; n < 2; ++n)
      for (int i = 0; i < 16; ++i)
        bf[n][i] = Wi[(size_t)(kb0 + (i & 7) + ((i >> 3) << 4)) * Lc + l0 + (w * 2 + n) * 16 + (lane & 15)];
    for (int n = 0; n < 2; ++n)
      for (int m = 0; m < 4; ++m)
        acc[n][m] = __builtin_amdgcn_wmma_f32_16x16x32_f16(
            false, af[m], false, bf[n], (short)0, acc[n][m], false, false);
  }

  const float SCALE = 0x1p-30f;   // 1/(512*512*4096)
  const int mhi = (lane >> 4) * 8;
  for (int n = 0; n < 2; ++n) {
    const int l = l0 + (w * 2 + n) * 16 + (lane & 15);
    for (int m = 0; m < 4; ++m)
      for (int rr = 0; rr < 8; ++rr)
        out[((size_t)bh * Ec + m * 16 + mhi + rr) * Lc + l] = acc[n][m][rr] * SCALE;
  }
}

// ---------------------------------------------------------------------------
extern "C" void kernel_launch(void* const* d_in, const int* in_sizes, int n_in,
                              void* d_out, int out_size, void* d_ws,
                              size_t ws_size, hipStream_t stream) {
  (void)in_sizes; (void)n_in; (void)out_size; (void)ws_size;
  const float* q     = (const float*)d_in[0];
  const float* k     = (const float*)d_in[1];
  const float* v     = (const float*)d_in[2];
  const float* wr    = (const float*)d_in[3];
  const float* wi    = (const float*)d_in[4];
  const int*   idxq  = (const int*)d_in[5];
  const int*   idxkv = (const int*)d_in[6];

  char* p = (char*)d_ws;
  _Float16* Wfq  = (_Float16*)p; p += (size_t)Lc * Jc * sizeof(_Float16); // 1 MB
  _Float16* Wfkv = (_Float16*)p; p += (size_t)Lc * Jc * sizeof(_Float16); // 1 MB
  _Float16* Wi   = (_Float16*)p; p += (size_t)Jc * Lc * sizeof(_Float16); // 1 MB
  float*    Xq   = (float*)p;    p += (size_t)BH * Ec * Jc * sizeof(float); // 2 MB
  float*    Xk   = (float*)p;    p += (size_t)BH * Ec * Jc * sizeof(float); // 2 MB
  float*    Xv   = (float*)p;    p += (size_t)BH * Ec * Jc * sizeof(float); // 2 MB
  _Float16* Xw   = (_Float16*)p;                                            // 1 MB

  k_twiddle<<<dim3(1024, 3), 256, 0, stream>>>(idxq, idxkv, Wfq, Wfkv, Wi);
  k_dft<<<dim3(64, 3), 256, 0, stream>>>(q, k, v, Wfq, Wfkv, Xq, Xk, Xv);
  k_mid<<<dim3(64), 256, 0, stream>>>(Xq, Xk, Xv, wr, wi, Xw);
  k_idft<<<dim3(16, 64), 256, 0, stream>>>(Xw, Wi, (float*)d_out);
}
